// PCA_PDBL_Head_41188736369365
// MI455X (gfx1250) — compile-verified, matrix operand order checked by reading, not run on hardware
//
#include <hip/hip_runtime.h>

// Problem dims (fixed by the reference)
#define NDIM 8192   // rows of feats / out
#define DDIM 8192   // feature dim (K of main GEMM)
#define KDIM 2048   // PCA components
#define CDIM 1000   // classes
#define CPAD 1024   // padded C (multiple of 128)

typedef __attribute__((ext_vector_type(16))) __bf16         v16bf;
typedef __attribute__((ext_vector_type(8)))  float          v8f;
typedef __attribute__((ext_vector_type(8)))  unsigned short ush8;
typedef __attribute__((ext_vector_type(4)))  unsigned short ush4;

union FragBf { ush8 u[2]; v16bf v; };

// bf16 round-to-nearest-even from f32
__device__ __forceinline__ unsigned short f32_to_bf16(float f) {
  unsigned int u = __float_as_uint(f);
  unsigned int r = 0x7FFFu + ((u >> 16) & 1u);
  return (unsigned short)((u + r) >> 16);
}

// Split f32 into bf16 hi + bf16 lo such that hi + lo ~= f to ~16 mantissa bits.
__device__ __forceinline__ void split_f32(float f, unsigned short& hi,
                                          unsigned short& lo) {
  unsigned short h = f32_to_bf16(f);
  float fh = __uint_as_float((unsigned int)h << 16);
  hi = h;
  lo = f32_to_bf16(f - fh);
}

__device__ __forceinline__ void split4(const float4 v, ush4& hi, ush4& lo) {
  unsigned short h0, l0, h1, l1, h2, l2, h3, l3;
  split_f32(v.x, h0, l0); split_f32(v.y, h1, l1);
  split_f32(v.z, h2, l2); split_f32(v.w, h3, l3);
  hi = (ush4){h0, h1, h2, h3};
  lo = (ush4){l0, l1, l2, l3};
}

// ---- CDNA5 async global->LDS (ASYNCcnt) helpers -------------------------
// Flat pointers to __shared__ carry the LDS byte offset in their low 32 bits
// (LDS aperture lives in addr[63:32]); VDST of the async op wants that offset.
__device__ __forceinline__ void async_load_b128(void* lds_ptr, const void* gptr) {
  unsigned lds_off = (unsigned)(size_t)lds_ptr;
  asm volatile("global_load_async_to_lds_b128 %0, %1, off"
               :: "v"(lds_off), "v"(gptr)
               : "memory");
}
__device__ __forceinline__ void wait_asynccnt0() {
  asm volatile("s_wait_asynccnt 0x0" ::: "memory");
}

// Load one 16x32 bf16 operand fragment from an LDS tile laid out
// [16-row][LSTR-stride bf16, K contiguous]. Matches CDNA5 16-bit A-matrix
// layout: lanes 0-15 -> rows, K chunks {kh*8..+7} and {16+kh*8..+7}.
#define LSTR 40  // padded row stride in elements (80B, 16B-aligned rows)
__device__ __forceinline__ void load_frag(const unsigned short* ls, int row0,
                                          int lane, FragBf& f) {
  const int lr = lane & 15;
  const int kh = lane >> 4;
  const unsigned short* base = ls + (row0 + lr) * LSTR + kh * 8;
  f.u[0] = *(const ush8*)(base);
  f.u[1] = *(const ush8*)(base + 16);
}

#define WMMA_BF16(A, B, C) \
  __builtin_amdgcn_wmma_f32_16x16x32_bf16(false, (A), false, (B), (short)0, (C), false, false)

// ---------------------------------------------------------------------------
// Kernel 1: t[k] = dot(mean, comp[k,:])
// ---------------------------------------------------------------------------
__global__ __launch_bounds__(256)
void mean_proj_kernel(const float* __restrict__ mean,
                      const float* __restrict__ comp,
                      float* __restrict__ t) {
  __shared__ float red[256];
  const int k = blockIdx.x;
  const float* row = comp + (size_t)k * DDIM;
  float s = 0.f;
  for (int d = threadIdx.x; d < DDIM; d += 256) s += mean[d] * row[d];
  red[threadIdx.x] = s;
  __syncthreads();
  for (int off = 128; off > 0; off >>= 1) {
    if (threadIdx.x < off) red[threadIdx.x] += red[threadIdx.x + off];
    __syncthreads();
  }
  if (threadIdx.x == 0) t[k] = red[0];
}

// ---------------------------------------------------------------------------
// Kernel 2: bias[c] = sum_k t[k] * W[k,c]   (bias[c>=CDIM] = 0)
// ---------------------------------------------------------------------------
__global__ __launch_bounds__(256)
void bias_kernel(const float* __restrict__ t,
                 const float* __restrict__ W,
                 float* __restrict__ bias) {
  const int c = blockIdx.x * 256 + threadIdx.x;
  if (c >= CPAD) return;
  float s = 0.f;
  if (c < CDIM) {
    for (int k = 0; k < KDIM; ++k) s += t[k] * W[(size_t)k * CDIM + c];
  }
  bias[c] = s;
}

// ---------------------------------------------------------------------------
// Kernel 3: M = comp^T @ W, emitted as split bf16 planes
//   Mth/Mtl : [CPAD][DDIM] bf16, M^T layout (row = class c, col = d)
// WMMA GEMM (M-dim=c, N-dim=d, K-dim=k), operands split hi/lo in LDS,
// 3 WMMAs per product (hh + hl + lh). Rows c>=CDIM are zero (A zero-filled).
// ---------------------------------------------------------------------------
__global__ __launch_bounds__(256)
void fuse_projection_kernel(const float* __restrict__ comp,   // [KDIM][DDIM]
                            const float* __restrict__ W,      // [KDIM][CDIM]
                            unsigned short* __restrict__ Mth, // hi plane
                            unsigned short* __restrict__ Mtl) // lo plane
{
  __shared__ __align__(16) unsigned short lsAhi[128 * LSTR]; // [c_local][k]
  __shared__ __align__(16) unsigned short lsAlo[128 * LSTR];
  __shared__ __align__(16) unsigned short lsBhi[128 * LSTR]; // [d_local][k]
  __shared__ __align__(16) unsigned short lsBlo[128 * LSTR];

  const int tid  = threadIdx.x;
  const int lane = tid & 31;
  const int wave = tid >> 5;
  const int wm   = wave >> 2;          // 0..1 : c-dim wave row
  const int wn   = wave & 3;           // 0..3 : d-dim wave col
  const int c0   = blockIdx.y * 128;
  const int d0   = blockIdx.x * 128;
  const bool cfull = (c0 + 128 <= CDIM);   // block-uniform tail predicate

  v8f acc[4][2];
  const v8f vzero = {0.f,0.f,0.f,0.f,0.f,0.f,0.f,0.f};
  #pragma unroll
  for (int m = 0; m < 4; ++m)
    #pragma unroll
    for (int n = 0; n < 2; ++n) acc[m][n] = vzero;

  #pragma unroll 1
  for (int k0 = 0; k0 < KDIM; k0 += 32) {
    // A = W^T tile: 32 k-rows x 128 c-cols, transposed into lsA*[c][k]
    #pragma unroll
    for (int i = 0; i < 4; ++i) {
      int idx = tid + i * 256;         // 0..1023
      int kr  = idx >> 5;              // 0..31
      int cl  = (idx & 31) * 4;        // 0..124
      int cg  = c0 + cl;
      const float* p = W + (size_t)(k0 + kr) * CDIM + cg;
      float4 v;
      if (cfull) {
        v = *(const float4*)p;
      } else {
        v.x = (cg + 0 < CDIM) ? p[0] : 0.f;
        v.y = (cg + 1 < CDIM) ? p[1] : 0.f;
        v.z = (cg + 2 < CDIM) ? p[2] : 0.f;
        v.w = (cg + 3 < CDIM) ? p[3] : 0.f;
      }
      ush4 hi, lo;
      split4(v, hi, lo);
      #pragma unroll
      for (int j = 0; j < 4; ++j) {
        lsAhi[(cl + j) * LSTR + kr] = hi[j];
        lsAlo[(cl + j) * LSTR + kr] = lo[j];
      }
    }
    // B = comp tile: 32 k-rows x 128 d-cols, transposed into lsB*[d][k]
    #pragma unroll
    for (int i = 0; i < 4; ++i) {
      int idx = tid + i * 256;
      int kr  = idx >> 5;
      int dl  = (idx & 31) * 4;
      const float4 v = *(const float4*)(comp + (size_t)(k0 + kr) * DDIM + d0 + dl);
      ush4 hi, lo;
      split4(v, hi, lo);
      #pragma unroll
      for (int j = 0; j < 4; ++j) {
        lsBhi[(dl + j) * LSTR + kr] = hi[j];
        lsBlo[(dl + j) * LSTR + kr] = lo[j];
      }
    }
    __syncthreads();

    FragBf bh[2], bl[2];
    #pragma unroll
    for (int n = 0; n < 2; ++n) {
      load_frag(lsBhi, wn * 32 + n * 16, lane, bh[n]);
      load_frag(lsBlo, wn * 32 + n * 16, lane, bl[n]);
    }
    #pragma unroll
    for (int m = 0; m < 4; ++m) {
      FragBf ah, al;
      load_frag(lsAhi, wm * 64 + m * 16, lane, ah);
      load_frag(lsAlo, wm * 64 + m * 16, lane, al);
      #pragma unroll
      for (int n = 0; n < 2; ++n) {
        acc[m][n] = WMMA_BF16(ah.v, bh[n].v, acc[m][n]);  // hi*hi
        acc[m][n] = WMMA_BF16(ah.v, bl[n].v, acc[m][n]);  // hi*lo
        acc[m][n] = WMMA_BF16(al.v, bh[n].v, acc[m][n]);  // lo*hi
      }
    }
    __syncthreads();
  }

  // Epilogue: split f32 acc -> (Mth, Mtl)
  const int lr = lane & 15;
  const int hl = lane >> 4;
  #pragma unroll
  for (int m = 0; m < 4; ++m) {
    #pragma unroll
    for (int n = 0; n < 2; ++n) {
      const int d = d0 + wn * 32 + n * 16 + lr;
      #pragma unroll
      for (int i = 0; i < 8; ++i) {
        const int c = c0 + wm * 64 + m * 16 + hl * 8 + i;
        unsigned short hi, lo;
        split_f32(acc[m][n][i], hi, lo);
        Mth[(size_t)c * DDIM + d] = hi;
        Mtl[(size_t)c * DDIM + d] = lo;
      }
    }
  }
}

// ---------------------------------------------------------------------------
// Kernel 4: out[n][c] = sum_d feats[n,d]*M[d,c] - bias[c]
// Block tile 128x128, BK=32, 8 waves of 64x32 -> 24 WMMA / k-step (bf16x3).
// A (feats) is register-staged (needs fp32->bf16 hi/lo split in VALU);
// B (bf16 Mt planes) streams via GLOBAL_LOAD_ASYNC_TO_LDS_B128 into a
// double-buffered LDS tile, synchronized with s_wait_asynccnt + barrier.
// Mt planes (32 MB) are L2-resident (192 MB); HBM streams feats once.
// ---------------------------------------------------------------------------
__global__ __launch_bounds__(256)
void main_gemm_kernel(const float* __restrict__ feats,           // [NDIM][DDIM]
                      const unsigned short* __restrict__ Mth,    // [CPAD][DDIM]
                      const unsigned short* __restrict__ Mtl,    // [CPAD][DDIM]
                      const float* __restrict__ bias,            // [CPAD]
                      float* __restrict__ out)                   // [NDIM][CDIM]
{
  __shared__ __align__(16) unsigned short lsAhi[128 * LSTR];     // [n_local][k]
  __shared__ __align__(16) unsigned short lsAlo[128 * LSTR];
  __shared__ __align__(16) unsigned short lsBhi[2][128 * LSTR];  // [buf][c_local][k]
  __shared__ __align__(16) unsigned short lsBlo[2][128 * LSTR];

  const int tid  = threadIdx.x;
  const int lane = tid & 31;
  const int wave = tid >> 5;
  const int wm   = wave >> 2;          // 0..1 : n-dim wave row
  const int wn   = wave & 3;           // 0..3 : c-dim wave col
  const int n0   = blockIdx.y * 128;
  const int cb0  = blockIdx.x * 128;

  v8f acc[4][2];
  const v8f vzero = {0.f,0.f,0.f,0.f,0.f,0.f,0.f,0.f};
  #pragma unroll
  for (int m = 0; m < 4; ++m)
    #pragma unroll
    for (int n = 0; n < 2; ++n) acc[m][n] = vzero;

  // A-operand register staging
  float4 ra[4];
  auto load_A_regs = [&](int dd) {
    #pragma unroll
    for (int i = 0; i < 4; ++i) {
      int idx = tid + i * 256;         // 0..1023
      int r   = idx >> 3;              // 0..127
      int q   = idx & 7;               // float4 slot
      const float* p = feats + (size_t)(n0 + r) * DDIM + dd + q * 4;
      ra[i] = *(const float4*)p;
      if (dd + 32 < DDIM) __builtin_prefetch(p + 32, 0, 0);  // 2 stages ahead
    }
  };

  // B-operand: issue async global->LDS copies (4 x b128 per thread)
  auto issue_B_async = [&](int dd, int buf) {
    #pragma unroll
    for (int i = 0; i < 2; ++i) {
      int idx = tid + i * 256;         // 0..511
      int r   = idx >> 2;              // 0..127
      int q   = idx & 3;               // ush8 slot
      size_t goff = (size_t)(cb0 + r) * DDIM + dd + q * 8;
      int    loff = r * LSTR + q * 8;
      async_load_b128(&lsBhi[buf][loff], Mth + goff);
      async_load_b128(&lsBlo[buf][loff], Mtl + goff);
    }
  };

  // Pipeline prologue: stage 0 in flight
  issue_B_async(0, 0);
  load_A_regs(0);

  #pragma unroll 1
  for (int d0 = 0; d0 < DDIM; d0 += 32) {
    const int cur = (d0 >> 5) & 1;
    __syncthreads();   // all waves done reading LDS from previous stage
    // Commit staged A registers to LDS (fp32 -> bf16 hi/lo)
    #pragma unroll
    for (int i = 0; i < 4; ++i) {
      int idx = tid + i * 256;
      int r   = idx >> 3;
      int q   = idx & 7;
      ush4 hi, lo;
      split4(ra[i], hi, lo);
      *(ush4*)&lsAhi[r * LSTR + q * 4] = hi;
      *(ush4*)&lsAlo[r * LSTR + q * 4] = lo;
    }
    wait_asynccnt0();                      // this stage's B tile landed in LDS
    if (d0 + 32 < DDIM) issue_B_async(d0 + 32, cur ^ 1);  // WAR-safe after barrier
    __syncthreads();                       // publish A stores + B async writes

    if (d0 + 32 < DDIM) load_A_regs(d0 + 32);  // overlaps the WMMAs below

    FragBf bh[2], bl[2];
    #pragma unroll
    for (int n = 0; n < 2; ++n) {
      load_frag(lsBhi[cur], wn * 32 + n * 16, lane, bh[n]);
      load_frag(lsBlo[cur], wn * 32 + n * 16, lane, bl[n]);
    }
    #pragma unroll
    for (int m = 0; m < 4; ++m) {
      FragBf ah, al;
      load_frag(lsAhi, wm * 64 + m * 16, lane, ah);
      load_frag(lsAlo, wm * 64 + m * 16, lane, al);
      #pragma unroll
      for (int n = 0; n < 2; ++n) {
        acc[m][n] = WMMA_BF16(ah.v, bh[n].v, acc[m][n]);  // hi*hi
        acc[m][n] = WMMA_BF16(ah.v, bl[n].v, acc[m][n]);  // hi*lo
        acc[m][n] = WMMA_BF16(al.v, bh[n].v, acc[m][n]);  // lo*hi
      }
    }
  }

  // Epilogue: subtract bias, guarded store (CDIM = 1000 < CPAD)
  const int lr = lane & 15;
  const int hl = lane >> 4;
  #pragma unroll
  for (int m = 0; m < 4; ++m) {
    #pragma unroll
    for (int n = 0; n < 2; ++n) {
      const int c = cb0 + wn * 32 + n * 16 + lr;
      const float bv = bias[c];
      if (c < CDIM) {
        #pragma unroll
        for (int i = 0; i < 8; ++i) {
          const int nr = n0 + wm * 64 + m * 16 + hl * 8 + i;
          out[(size_t)nr * CDIM + c] = acc[m][n][i] - bv;
        }
      }
    }
  }
}

// ---------------------------------------------------------------------------
extern "C" void kernel_launch(void* const* d_in, const int* in_sizes, int n_in,
                              void* d_out, int out_size, void* d_ws, size_t ws_size,
                              hipStream_t stream) {
  const float* feats = (const float*)d_in[0];  // [N,D]
  const float* mean  = (const float*)d_in[1];  // [D]
  const float* comp  = (const float*)d_in[2];  // [K,D]
  const float* W     = (const float*)d_in[3];  // [K,C]
  float* out         = (float*)d_out;          // [N,C]

  // Workspace: Mth | Mtl (bf16 [CPAD][DDIM] each) | t (f32[K]) | bias (f32[CPAD])
  const size_t planeElems = (size_t)CPAD * DDIM;
  unsigned short* Mth = (unsigned short*)d_ws;
  unsigned short* Mtl = Mth + planeElems;
  float* t    = (float*)(Mtl + planeElems);
  float* bias = t + KDIM;

  mean_proj_kernel<<<KDIM, 256, 0, stream>>>(mean, comp, t);
  bias_kernel<<<CPAD / 256, 256, 0, stream>>>(t, W, bias);
  fuse_projection_kernel<<<dim3(DDIM / 128, CPAD / 128), 256, 0, stream>>>(comp, W, Mth, Mtl);
  main_gemm_kernel<<<dim3(CPAD / 128, NDIM / 128), 256, 0, stream>>>(feats, Mth, Mtl, bias, out);
}
